// QuantLinearChannelWise_17377437680105
// MI455X (gfx1250) — compile-verified, matrix operand order checked by reading, not run on hardware
//
#include <hip/hip_runtime.h>
#include <hip/hip_bf16.h>

// QuantLinearChannelWise on gfx1250 (MI455X):
//   out[M,N] = (x[M,K] @ (q[N,K])^T) * scale[N] + bias[N]
// Scale folded into the epilogue; GEMM core runs V_WMMA_F32_16X16X32_F16
// (int8 weights are exact in f16, x rounded RTN, f32 accumulation).
// M=16384 (B*S), N=11008, K=4096 -> compute-bound (1.48 TFLOP).
//
// Pipeline per K-step: issue next tile's global_load_b128 into registers,
// run ds_load_b128 + 8x WMMA on the current LDS buffer (hides HBM latency),
// then wait/convert/ds_store into the other buffer, one barrier.

typedef __attribute__((ext_vector_type(16))) _Float16 v16h;
typedef __attribute__((ext_vector_type(8)))  _Float16 v8h;
typedef __attribute__((ext_vector_type(4)))  _Float16 v4h;
typedef __attribute__((ext_vector_type(8)))  float    v8f;
typedef __attribute__((ext_vector_type(4)))  float    v4f;
typedef __attribute__((ext_vector_type(4)))  int      v4i;

#define TM 128          // block tile M
#define TN 128          // block tile N
#define TK 32           // K step (one WMMA K)
#define LDT 40          // padded LDS row stride in halves (conflict-free b128 reads)
#define NTHREADS 256    // 8 waves (wave32)

__global__ __launch_bounds__(NTHREADS)
void qlinear_wmma_f16_kernel(const float* __restrict__ x,
                             const int*   __restrict__ qw,
                             const float* __restrict__ scale,
                             const float* __restrict__ bias,
                             float*       __restrict__ out,
                             int N, int K)
{
    __shared__ _Float16 Ash[2][TM * LDT];   // x tile, f16
    __shared__ _Float16 Bsh[2][TN * LDT];   // weight tile, f16 (exact int8)

    const int t    = threadIdx.x;
    const int lane = t & 31;
    const int wave = t >> 5;
    const int lrow = lane & 15;     // row/col within 16x16 subtile
    const int lhal = lane >> 4;     // lane half selects K-chunk per ISA A-layout

    const int bm = blockIdx.y * TM;
    const int bn = blockIdx.x * TN;
    const int wm = (wave & 3) * 32;  // 4 waves along M
    const int wn = (wave >> 2) * 64; // 2 waves along N

    const int nk = K / TK;

    // Per-thread tile-chunk coordinates (4 x float4 per tile per matrix).
    int rowv[4], c4v[4];
    #pragma unroll
    for (int i = 0; i < 4; ++i) {
        const int f = t + NTHREADS * i;     // 0..1023 float4 chunks
        rowv[i] = f >> 3;                   // 0..127
        c4v[i]  = (f & 7) * 4;              // 0,4,..,28
    }

    v8f acc[2][4];
    #pragma unroll
    for (int mi = 0; mi < 2; ++mi)
        #pragma unroll
        for (int ni = 0; ni < 4; ++ni)
            acc[mi][ni] = {};

    // ---- phase 1: issue global loads for K-tile kk into registers ----
    v4f areg[4];
    v4i breg[4];
    auto gload = [&](int kk) {
        const int k0 = kk * TK;
        #pragma unroll
        for (int i = 0; i < 4; ++i) {
            areg[i] = *(const v4f*)(x  + (size_t)(bm + rowv[i]) * K + k0 + c4v[i]);
            breg[i] = *(const v4i*)(qw + (size_t)(bn + rowv[i]) * K + k0 + c4v[i]);
        }
    };

    // ---- phase 2: convert staged registers and store to LDS buffer ----
    auto lstore = [&](int buf) {
        #pragma unroll
        for (int i = 0; i < 4; ++i) {
            v4h ah = { (_Float16)areg[i].x, (_Float16)areg[i].y,
                       (_Float16)areg[i].z, (_Float16)areg[i].w };
            *(v4h*)&Ash[buf][rowv[i] * LDT + c4v[i]] = ah;
            v4h bh = { (_Float16)(float)breg[i].x, (_Float16)(float)breg[i].y,
                       (_Float16)(float)breg[i].z, (_Float16)(float)breg[i].w };
            *(v4h*)&Bsh[buf][rowv[i] * LDT + c4v[i]] = bh;
        }
    };

    // Build a 16x32 f16 fragment per the ISA A/B VGPR layout:
    // lanes 0-15 hold K 0..7 & 16..23, lanes 16-31 hold K 8..15 & 24..31.
    auto frag16 = [&](const _Float16* base) -> v16h {
        v8h lo = *(const v8h*)(base);        // ds_load_b128
        v8h hi = *(const v8h*)(base + 16);   // ds_load_b128 (K+16 chunk)
        return __builtin_shufflevector(lo, hi,
                0,1,2,3,4,5,6,7,8,9,10,11,12,13,14,15);
    };

    // Prologue: fill buffer 0.
    gload(0);
    lstore(0);
    __syncthreads();

    for (int kk = 0; kk < nk; ++kk) {
        const int  cur      = kk & 1;
        const bool havenext = (kk + 1 < nk);

        // Issue next tile's global loads; do NOT consume them yet.
        if (havenext)
            gload(kk + 1);

        if (kk + 2 < nk) {                  // global_prefetch_b8, 2 tiles ahead
            __builtin_prefetch(x  + (size_t)(bm + rowv[0]) * K + (kk + 2) * TK + c4v[0], 0, 1);
            __builtin_prefetch(qw + (size_t)(bn + rowv[0]) * K + (kk + 2) * TK + c4v[0], 0, 1);
        }

        // Matrix work on the current buffer (global loads in flight).
        v16h af[2], bf[4];
        #pragma unroll
        for (int mi = 0; mi < 2; ++mi)
            af[mi] = frag16(&Ash[cur][(wm + mi * 16 + lrow) * LDT + lhal * 8]);
        #pragma unroll
        for (int ni = 0; ni < 4; ++ni)
            bf[ni] = frag16(&Bsh[cur][(wn + ni * 16 + lrow) * LDT + lhal * 8]);

        #pragma unroll
        for (int mi = 0; mi < 2; ++mi)
            #pragma unroll
            for (int ni = 0; ni < 4; ++ni)
                acc[mi][ni] = __builtin_amdgcn_wmma_f32_16x16x32_f16(
                    /*neg_a=*/false, af[mi],
                    /*neg_b=*/false, bf[ni],
                    /*c_mod=*/(short)0, acc[mi][ni],
                    /*reuse_a=*/false, /*reuse_b=*/false);

        // Now consume the staged loads: convert + store into the other buffer.
        if (havenext)
            lstore(cur ^ 1);

        __syncthreads();
    }

    // Epilogue: out = acc * scale[col] + bias[col].
    // C/D layout: VGPR i -> row i (lanes 0-15) / row i+8 (lanes 16-31), col = lane&15.
    #pragma unroll
    for (int ni = 0; ni < 4; ++ni) {
        const int col = bn + wn + ni * 16 + lrow;
        const float s = scale[col];
        const float b = bias[col];
        #pragma unroll
        for (int mi = 0; mi < 2; ++mi) {
            const int rbase = bm + wm + mi * 16 + lhal * 8;
            #pragma unroll
            for (int i = 0; i < 8; ++i)
                out[(size_t)(rbase + i) * N + col] = acc[mi][ni][i] * s + b;
        }
    }
}

extern "C" void kernel_launch(void* const* d_in, const int* in_sizes, int n_in,
                              void* d_out, int out_size, void* d_ws, size_t ws_size,
                              hipStream_t stream) {
    const float* x     = (const float*)d_in[0];   // [B,S,IN] f32
    const int*   qw    = (const int*)  d_in[1];   // [OUT,IN] int32 (int8-range)
    const float* scale = (const float*)d_in[2];   // [OUT]
    const float* bias  = (const float*)d_in[3];   // [OUT]
    float*       out   = (float*)d_out;           // [B,S,OUT]

    const int K = 4096;
    const int N = in_sizes[2];          // 11008
    const int M = in_sizes[0] / K;      // 8*2048 = 16384

    dim3 grid((unsigned)(N / TN), (unsigned)(M / TM));
    qlinear_wmma_f16_kernel<<<grid, NTHREADS, 0, stream>>>(x, qw, scale, bias, out, N, K);
}
